// MyLSTMModel_66391604462338
// MI455X (gfx1250) — matrix operand order repaired and checked by reading, not compiled, not run on hardware
//
#include <hip/hip_runtime.h>
#include <hip/hip_bf16.h>
#include <math.h>

typedef __attribute__((ext_vector_type(16))) __bf16 v16bf;
typedef __attribute__((ext_vector_type(8)))  __bf16 v8bf;
typedef __attribute__((ext_vector_type(8)))  float  v8f;

#define BB   16
#define TT   16
#define HH   161
#define WW   181
#define CIN  3
#define FF   21            // ConvLSTM filters
#define F4   84            // 4*FF gate channels
#define NPOS (BB*HH*WW)    // 466256 spatial positions
#define PC   24            // packed channels: 3 x-channels + 21 h-channels
#define KTOT 216           // 9 taps * 24 channels
#define KS   7             // K-steps of 32 (K padded 216 -> 224)
#define NT   6             // N tiles of 16 (84 padded -> 96)
#define ZSTR 96            // z row stride (channels, padded)
#define LDA  232           // LDS A-tile row stride in halfs (224 + 8 pad, bank-friendly)

// workspace layout (bytes); total ~240.6 MB
#define SZ_WB  ((size_t)KS*NT*32*16*2)       // 43008  : pre-swizzled bf16 weights
#define OFF_P  ((size_t)43008)
#define SZ_P   ((size_t)NPOS*PC*2)           // 22380288 : packed bf16 input [x|h]
#define OFF_C  (OFF_P + SZ_P)
#define SZ_C   ((size_t)NPOS*FF*4)           // 39165504 : fp32 cell state
#define OFF_Z  (OFF_C + SZ_C)                // z: NPOS*96*4 = 179042304

__global__ void zero_u32_kernel(unsigned* __restrict__ p, long n) {
  long i = (long)blockIdx.x * blockDim.x + threadIdx.x;
  if (i < n) p[i] = 0u;
}

// Pre-swizzle [Wx;Wh] into exact WMMA B-fragment order:
// wB[((s*NT + nt)*32 + lane)*16 + e]  holds  W[K = 32s + 8*(lane>>4) + (e&7) + 16*(e>>3), N = nt*16 + (lane&15)]
// (16-bit B 32x16: lanes 0-15 hold K=0..7,16..23, lanes 16-31 hold K=8..15,24..31, 2 halfs per VGPR)
__global__ void packW_kernel(const float* __restrict__ Wx, const float* __restrict__ Wh,
                             __bf16* __restrict__ wB) {
  int i = blockIdx.x * 256 + threadIdx.x;
  if (i >= KS * NT * 32 * 16) return;
  int e    = i & 15;
  int lane = (i >> 4) & 31;
  int nt   = (i >> 9) % NT;
  int s    = i / (16 * 32 * NT);
  int N = nt * 16 + (lane & 15);
  int K = 32 * s + 8 * (lane >> 4) + (e & 7) + 16 * (e >> 3);
  float v = 0.f;
  if (K < KTOT && N < F4) {
    int tap = K / PC, c = K - tap * PC;            // tap = ky*3+kx
    v = (c < CIN) ? Wx[(tap * CIN + c) * F4 + N]
                  : Wh[(tap * FF + (c - CIN)) * F4 + N];
  }
  wB[i] = (__bf16)v;
}

// Pack frame t of x into channels 0..2 of the bf16 packed input (channels 3..23 = h, written by gates).
__global__ void pack_x_kernel(const float* __restrict__ x, __bf16* __restrict__ P, int t) {
  int i = blockIdx.x * 256 + threadIdx.x;
  if (i >= NPOS) return;
  int b = i / (HH * WW);
  int r = i - b * (HH * WW);
  const float* xp = x + (((size_t)b * TT + t) * (HH * WW) + r) * CIN;
  __bf16* pp = P + (size_t)i * PC;
  pp[0] = (__bf16)xp[0];
  pp[1] = (__bf16)xp[1];
  pp[2] = (__bf16)xp[2];
}

// Implicit-GEMM 3x3 SAME conv over the 24-channel packed input -> z[...,96] (f32).
// Block: 384 threads = 12 waves = 2 M-tiles(16) x 6 N-tiles(16); M = 32 positions along W.
__global__ void __launch_bounds__(384)
convgemm_kernel(const __bf16* __restrict__ P, const __bf16* __restrict__ wB,
                float* __restrict__ z) {
  __shared__ __bf16 As[32 * LDA];                 // 32 x 224 im2col tile (padded stride)
  const int b  = blockIdx.z;
  const int y  = blockIdx.y;
  const int x0 = blockIdx.x * 32;
  const int tid = threadIdx.x;

  // im2col fill at b64 granularity: 32 rows x 56 half-quads.
  // Channel quads never straddle a tap (24 % 4 == 0); all addresses 8B-aligned.
  for (int idx = tid; idx < 32 * 56; idx += 384) {
    int m = idx / 56;
    int k = (idx - m * 56) * 4;
    unsigned long long val = 0ull;
    if (k < KTOT) {
      int tap = k / PC, c = k - tap * PC;
      int yy = y + (tap / 3) - 1;
      int xx = x0 + m + (tap % 3) - 1;
      if (yy >= 0 && yy < HH && xx >= 0 && xx < WW)
        val = *(const unsigned long long*)(P + (((size_t)b * HH + yy) * WW + xx) * PC + c);
    }
    *(unsigned long long*)(&As[m * LDA + k]) = val;
  }
  __syncthreads();

  const int wave = tid >> 5, lane = tid & 31;
  const int msub = wave & 1, nt = wave >> 1;
  const int hi16 = lane >> 4;
  // ISA 16-bit A 16x32 layout: lane holds row M=lane&15; halfs 0..7 at K=8*(lane>>4)+i,
  // halfs 8..15 at K=8*(lane>>4)+16+i  -> two contiguous 16B chunks per lane.
  const __bf16* arow = &As[(msub * 16 + (lane & 15)) * LDA + 8 * hi16];
  const __bf16* bptr = wB + ((size_t)nt * 32 + lane) * 16;

  v8f acc = {};
#pragma unroll
  for (int s = 0; s < KS; ++s) {
    v8bf lo = *(const v8bf*)(arow + 32 * s);
    v8bf hi = *(const v8bf*)(arow + 32 * s + 16);
    v16bf a;
#pragma unroll
    for (int i = 0; i < 8; ++i) { a[i] = lo[i]; a[i + 8] = hi[i]; }
    v16bf bf = *(const v16bf*)(bptr + (size_t)s * (NT * 32 * 16));
    acc = __builtin_amdgcn_wmma_f32_16x16x32_bf16(false, a, false, bf,
                                                  (short)0, acc, false, false);
  }

  // C/D layout: col N = lane&15; VGPR v -> row M = v + 8*(lane>>4)
  const int N = nt * 16 + (lane & 15);
  const int mbase = msub * 16 + 8 * hi16;
  float* zp = z + (((size_t)b * HH + y) * WW + (x0 + mbase)) * ZSTR + N;
  if (x0 + 32 <= WW) {
    // full tile (5 of 6 x-blocks): unconditional stores, constant offsets off one base
#pragma unroll
    for (int v = 0; v < 8; ++v) zp[v * ZSTR] = acc[v];
  } else {
#pragma unroll
    for (int v = 0; v < 8; ++v)
      if (x0 + mbase + v < WW) zp[v * ZSTR] = acc[v];
  }
}

// Gate math (Keras order i,f,c,o; hard_sigmoid), update c, write bf16 h back into packed
// input for next step; on last step apply mask and emit fp32 output.
__global__ void gates_kernel(const float* __restrict__ z, float* __restrict__ c,
                             __bf16* __restrict__ P, const float* __restrict__ bias,
                             const float* __restrict__ mask, float* __restrict__ out,
                             int last) {
  long i = (long)blockIdx.x * 256 + threadIdx.x;
  if (i >= (long)NPOS * FF) return;
  int pos = (int)(i / FF);
  int f   = (int)(i - (long)pos * FF);
  const float* zp = z + (size_t)pos * ZSTR;
  float zi = zp[f]          + bias[f];
  float zf = zp[FF + f]     + bias[FF + f];
  float zg = zp[2 * FF + f] + bias[2 * FF + f];
  float zo = zp[3 * FF + f] + bias[3 * FF + f];
  float ig = fminf(fmaxf(0.2f * zi + 0.5f, 0.f), 1.f);
  float fg = fminf(fmaxf(0.2f * zf + 0.5f, 0.f), 1.f);
  float og = fminf(fmaxf(0.2f * zo + 0.5f, 0.f), 1.f);
  float cn = fg * c[i] + ig * tanhf(zg);
  float h  = og * tanhf(cn);
  c[i] = cn;
  P[(size_t)pos * PC + CIN + f] = (__bf16)h;
  if (last) out[i] = h * mask[pos % (HH * WW)];
}

extern "C" void kernel_launch(void* const* d_in, const int* in_sizes, int n_in,
                              void* d_out, int out_size, void* d_ws, size_t ws_size,
                              hipStream_t stream) {
  (void)in_sizes; (void)n_in; (void)out_size; (void)ws_size;
  const float* x    = (const float*)d_in[0];   // (16,16,161,181,3)
  const float* Wx   = (const float*)d_in[1];   // (3,3,3,84)
  const float* Wh   = (const float*)d_in[2];   // (3,3,21,84)
  const float* bias = (const float*)d_in[3];   // (84,)
  const float* mask = (const float*)d_in[4];   // (161,181,1)
  float* out = (float*)d_out;                  // (16*161*181*21,) fp32

  char* ws = (char*)d_ws;
  __bf16* wB = (__bf16*)ws;
  __bf16* P  = (__bf16*)(ws + OFF_P);
  float*  cS = (float*)(ws + OFF_C);
  float*  zS = (float*)(ws + OFF_Z);

  // zero packed-input (h slots) and cell state
  {
    long nP = (long)(SZ_P / 4);
    long nC = (long)(SZ_C / 4);
    zero_u32_kernel<<<(unsigned)((nP + 255) / 256), 256, 0, stream>>>((unsigned*)P, nP);
    zero_u32_kernel<<<(unsigned)((nC + 255) / 256), 256, 0, stream>>>((unsigned*)cS, nC);
  }
  // pre-swizzle weights into WMMA B-fragment layout (once)
  packW_kernel<<<(KS * NT * 32 * 16 + 255) / 256, 256, 0, stream>>>(Wx, Wh, wB);

  dim3 ggemm((WW + 31) / 32, HH, BB);          // (6,161,16)
  const unsigned gpos   = (NPOS + 255) / 256;
  const unsigned ggates = (unsigned)(((long)NPOS * FF + 255) / 256);

  for (int t = 0; t < TT; ++t) {
    pack_x_kernel<<<gpos, 256, 0, stream>>>(x, P, t);
    convgemm_kernel<<<ggemm, 384, 0, stream>>>(P, wB, zS);
    gates_kernel<<<ggates, 256, 0, stream>>>(zS, cS, P, bias, mask, out,
                                             (t == TT - 1) ? 1 : 0);
  }
}
